// AutonomousSystem_89644557402406
// MI455X (gfx1250) — matrix operand order repaired and checked by reading, not compiled, not run on hardware
//
#include <hip/hip_runtime.h>
#include <math.h>

typedef __attribute__((ext_vector_type(16))) _Float16 v16h;
typedef __attribute__((ext_vector_type(8)))  _Float16 v8h;
typedef __attribute__((ext_vector_type(8)))  float    v8f;

#define MDIM   256
#define NQ_    64
#define TILE_M 64
#define BATCH_ 524288
#define NTHR   512
#define SCALE_F 0.41887902047863906f   // 2*pi/15

// Padded LDS row strides: stride(dwords) mod 64 == 4 or 36 -> conflict-free
#define XO_S 260   // sXout row stride (floats):  260 dw % 64 = 4
#define A_S  264   // sA    row stride (halves):  132 dw % 64 = 4
#define W_S  72    // sW    row stride (halves):   36 dw % 64 = 36

__device__ __forceinline__ float fast_tanh(float v) {
#if __has_builtin(__builtin_amdgcn_tanhf)
  return __builtin_amdgcn_tanhf(v);
#elif __has_builtin(__builtin_amdgcn_tanh_f32)
  return __builtin_amdgcn_tanh_f32(v);
#else
  return tanhf(v);
#endif
}

// ---------------------------------------------------------------------------
// eta: det of each SU(2) op, mean angle / pi  (64 elements -> one tiny block)
// ---------------------------------------------------------------------------
__global__ void eta_kernel(const int* __restrict__ qp, float* __restrict__ eta_out) {
  __shared__ float red[NQ_];
  const int q = threadIdx.x;
  float a = qp[q * 3 + 0] * SCALE_F;
  float b = qp[q * 3 + 1] * SCALE_F;
  float g = qp[q * 3 + 2] * SCALE_F;
  float cb = cosf(0.5f * b), sb = sinf(0.5f * b);
  float cp = cosf(0.5f * (a + g)), sp = sinf(0.5f * (a + g));
  float cm = cosf(0.5f * (a - g)), sm = sinf(0.5f * (a - g));
  float u00r = cp * cb, u00i = sp * cb;
  float u01r = cm * sb, u01i = sm * sb;
  float u10r = -cm * sb, u10i = sm * sb;
  float u11r = cp * cb, u11i = -sp * cb;
  float dr = (u00r * u11r - u00i * u11i) - (u01r * u10r - u01i * u10i);
  float di = (u00r * u11i + u00i * u11r) - (u01r * u10i + u01i * u10r);
  red[q] = atan2f(di, dr);
  __syncthreads();
  if (q == 0) {
    float s = 0.0f;
    for (int i = 0; i < NQ_; ++i) s += red[i];
    *eta_out = (s / (float)NQ_) * (float)(1.0 / M_PI);
  }
}

// ---------------------------------------------------------------------------
// Fused: SU(2) rotate -> f16 WMMA GEMM (x_out @ W^T) -> tanh(+bias) -> +x_out
// Block: 512 thr = 16 waves (4 row-waves x 4 col-waves). Tile M=64 x N=256.
// K streamed in 4 panels of 64, panel p+1 register-prefetched during compute.
// ---------------------------------------------------------------------------
__global__ __launch_bounds__(NTHR) void fused_su2_gemm(
    const float* __restrict__ x, const int* __restrict__ qp,
    const float* __restrict__ W, const float* __restrict__ bias,
    float* __restrict__ out)
{
  __shared__ float sS[NQ_ * 8];                            //  2 KB
  __shared__ __align__(16) float    sXout[TILE_M * XO_S];  // 65 KB fp32 residual
  __shared__ __align__(16) _Float16 sA[TILE_M * A_S];      // 33 KB f16 A tile
  __shared__ __align__(16) _Float16 sW[MDIM * W_S];        // 36 KB f16 W panel

  const int tid   = threadIdx.x;
  const int lane  = tid & 31;
  const int wave  = tid >> 5;            // 0..15
  const int mwave = (wave >> 2) * 16;    // 4 row-waves
  const int nwave = (wave & 3) * 64;     // 4 col-waves
  const size_t row0 = (size_t)blockIdx.x * TILE_M;

  // --- prefetch W panel 0 into registers (8 x float4 per thread) ---
  float4 wreg[8];
  #pragma unroll
  for (int i = 0; i < 8; ++i) {
    int id = tid + NTHR * i;             // 4096 tasks = 256 n x 16 chunks of 4
    int n = id >> 4, ch = (id & 15) * 4;
    wreg[i] = *(const float4*)(W + n * MDIM + ch);
  }

  // --- SU(2) operators (threads 0..63) ---
  if (tid < NQ_) {
    float a = qp[tid * 3 + 0] * SCALE_F;
    float b = qp[tid * 3 + 1] * SCALE_F;
    float g = qp[tid * 3 + 2] * SCALE_F;
    float cb = cosf(0.5f * b), sb = sinf(0.5f * b);
    float cp = cosf(0.5f * (a + g)), sp = sinf(0.5f * (a + g));
    float cm = cosf(0.5f * (a - g)), sm = sinf(0.5f * (a - g));
    float* Sq = sS + tid * 8;
    Sq[0] =  cp * cb; Sq[1] =  sp * cb;   // u00
    Sq[2] =  cm * sb; Sq[3] =  sm * sb;   // u01
    Sq[4] = -cm * sb; Sq[5] =  sm * sb;   // u10
    Sq[6] =  cp * cb; Sq[7] = -sp * cb;   // u11
  }
  __syncthreads();

  // --- rotate tile of x into sXout (fp32 residual) + sA (f16 GEMM operand) ---
  #pragma unroll
  for (int i = 0; i < 8; ++i) {
    int id  = tid + NTHR * i;            // 4096 tasks = 64 rows x 64 q-groups
    int row = id >> 6;
    int q   = id & 63;
    const float4 xv = *(const float4*)(x + (row0 + row) * MDIM + q * 4);
    const float* Sq = sS + q * 8;
    float y0r = Sq[0] * xv.x - Sq[1] * xv.y + Sq[2] * xv.z - Sq[3] * xv.w;
    float y0i = Sq[0] * xv.y + Sq[1] * xv.x + Sq[2] * xv.w + Sq[3] * xv.z;
    float y1r = Sq[4] * xv.x - Sq[5] * xv.y + Sq[6] * xv.z - Sq[7] * xv.w;
    float y1i = Sq[4] * xv.y + Sq[5] * xv.x + Sq[6] * xv.w + Sq[7] * xv.z;
    *(float4*)(sXout + row * XO_S + q * 4) = make_float4(y0r, y0i, y1r, y1i);
    union { _Float16 h[4]; uint2 u; } pk;
    pk.h[0] = (_Float16)y0r; pk.h[1] = (_Float16)y0i;
    pk.h[2] = (_Float16)y1r; pk.h[3] = (_Float16)y1i;
    *(uint2*)(sA + row * A_S + q * 4) = pk.u;
  }
  // sA/sXout consumed only after the barrier inside the panel loop below.

  v8f acc[4];
  const v8f vzero = {};
  #pragma unroll
  for (int t = 0; t < 4; ++t) acc[t] = vzero;

  // --- K loop: 4 panels of 64, software-pipelined W staging ---
  for (int p = 0; p < 4; ++p) {
    // commit prefetched panel p: cvt f32->f16, store to sW[n][k]
    #pragma unroll
    for (int i = 0; i < 8; ++i) {
      int id = tid + NTHR * i;
      int n = id >> 4, ch = (id & 15) * 4;
      union { _Float16 h[4]; uint2 u; } pk;
      pk.h[0] = (_Float16)wreg[i].x; pk.h[1] = (_Float16)wreg[i].y;
      pk.h[2] = (_Float16)wreg[i].z; pk.h[3] = (_Float16)wreg[i].w;
      *(uint2*)(sW + n * W_S + ch) = pk.u;
    }
    __syncthreads();

    // issue global loads for panel p+1; they fly while WMMAs run on panel p
    if (p < 3) {
      #pragma unroll
      for (int i = 0; i < 8; ++i) {
        int id = tid + NTHR * i;
        int n = id >> 4, ch = (id & 15) * 4;
        wreg[i] = *(const float4*)(W + n * MDIM + (p + 1) * 64 + ch);
      }
    }

    #pragma unroll
    for (int s = 0; s < 2; ++s) {
      const int kglob = p * 64 + s * 32;
      // A fragment 16x32 f16: lanes 0-15 K=0..15, lanes 16-31 K=16..31 (contig)
      const int m  = mwave + (lane & 15);
      const int kb = kglob + ((lane & 16) ? 16 : 0);
      v8h alo = *(const v8h*)(sA + m * A_S + kb);
      v8h ahi = *(const v8h*)(sA + m * A_S + kb + 8);
      v16h A = __builtin_shufflevector(alo, ahi,
                                       0, 1, 2, 3, 4, 5, 6, 7,
                                       8, 9, 10, 11, 12, 13, 14, 15);
      // B fragment 32x16: lanes 0-15 K=0..15, lanes 16-31 K=16..31 (contig)
      const int kbb = s * 32 + ((lane & 16) ? 16 : 0);
      #pragma unroll
      for (int t = 0; t < 4; ++t) {
        const int n = nwave + t * 16 + (lane & 15);
        v16h B = *(const v16h*)(sW + n * W_S + kbb);
        acc[t] = __builtin_amdgcn_wmma_f32_16x16x32_f16(
            false, A, false, B, (short)0, acc[t], false, false);
      }
    }
    __syncthreads();   // WAR on sW before committing next panel
  }

  // --- epilogue: tanh(C + bias) + x_out ---
  #pragma unroll
  for (int t = 0; t < 4; ++t) {
    const int n  = nwave + t * 16 + (lane & 15);
    const float bn = bias[n];
    #pragma unroll
    for (int r = 0; r < 8; ++r) {
      const int m = mwave + r + ((lane & 16) ? 8 : 0);
      float v = acc[t][r] + bn;
      out[(row0 + m) * MDIM + n] = fast_tanh(v) + sXout[m * XO_S + n];
    }
  }
}

extern "C" void kernel_launch(void* const* d_in, const int* in_sizes, int n_in,
                              void* d_out, int out_size, void* d_ws, size_t ws_size,
                              hipStream_t stream) {
  (void)in_sizes; (void)n_in; (void)d_ws; (void)ws_size; (void)out_size;
  const float* x  = (const float*)d_in[0];
  const int*   qp = (const int*)d_in[1];
  const float* W  = (const float*)d_in[2];
  const float* b  = (const float*)d_in[3];
  float* out = (float*)d_out;

  eta_kernel<<<1, NQ_, 0, stream>>>(qp, out + (size_t)BATCH_ * MDIM);
  fused_su2_gemm<<<BATCH_ / TILE_M, NTHR, 0, stream>>>(x, qp, W, b, out);
}
